// Generator_83760452207363
// MI455X (gfx1250) — matrix operand order repaired and checked by reading, not compiled, not run on hardware
//
#include <hip/hip_runtime.h>
#include <hip/hip_bf16.h>

// ---- problem sizes ----
#define B_BATCH 8192
#define HID     64
#define SEQ     180
#define LAT     32

// ---- LDS row strides (bf16 elems); padded so row*stride*2 is 16B aligned
// and 16 consecutive rows map to distinct bank groups (stride/2 dwords, gcd 4, period 16)
#define WS 72    // w_hh rows   (192 x WS)
#define FS 104   // fc_w rows   (64 x FS) and [z|embed] staging (16 x FS)
#define HS 72    // h buffers   (2 x 16 x HS)

typedef __bf16 bf16_t;
typedef bf16_t v16bf __attribute__((ext_vector_type(16)));
typedef bf16_t v2bf  __attribute__((ext_vector_type(2)));
typedef float  v8f   __attribute__((ext_vector_type(8)));

#define WMMA_BF16(A, B, C) \
  __builtin_amdgcn_wmma_f32_16x16x32_bf16(false, (A), false, (B), (short)0, (C), false, false)

__device__ __forceinline__ unsigned short f2bf(float f) {
  unsigned u = __builtin_bit_cast(unsigned, f);
  return (unsigned short)((u + 0x7fffu + ((u >> 16) & 1u)) >> 16); // RNE
}

// Pack two f32 -> packed bf16 pair (low = a, high = b).
// Fast path: round-half-up on each value's bits, then one v_perm_b32 to gather
// the two high halfwords (3 VALU total; differs from RNE only on exact ties).
__device__ __forceinline__ unsigned pack2bf(float a, float b) {
#if __has_builtin(__builtin_amdgcn_cvt_pk_bf16_f32)
  v2bf p = __builtin_amdgcn_cvt_pk_bf16_f32(a, b);
  return __builtin_bit_cast(unsigned, p);
#else
  unsigned ua = __builtin_bit_cast(unsigned, a) + 0x8000u;
  unsigned ub = __builtin_bit_cast(unsigned, b) + 0x8000u;
  // result bytes (LSB..MSB) = {ua.b2, ua.b3, ub.b2, ub.b3} = {bf(a), bf(b)}
  return __builtin_amdgcn_perm(ub, ua, 0x07060302u);
#endif
}

__device__ __forceinline__ float tanh_fast(float x) {
#if __has_builtin(__builtin_amdgcn_tanhf)
  return __builtin_amdgcn_tanhf(x);
#else
  float y;
  asm volatile("v_tanh_f32 %0, %1" : "=v"(y) : "v"(x));
  return y;
#endif
}

__global__ __launch_bounds__(128) void gru_gen_kernel(
    const float* __restrict__ z, const int* __restrict__ labels,
    const float* __restrict__ embed_w, const float* __restrict__ fc_w,
    const float* __restrict__ fc_b, const float* __restrict__ w_hh,
    const float* __restrict__ b_ih, const float* __restrict__ b_hh,
    const float* __restrict__ out_w, const float* __restrict__ out_b,
    float* __restrict__ out)
{
  // LDS carve-out: 27648 + 13312 + 3328 + 4608 = 48896 bytes
  __shared__ __align__(16) unsigned char smem[48896];
  unsigned char* WHH = smem;               // 192 x WS bf16 (w_hh row-major)
  unsigned char* FCW = smem + 27648;       // 64  x FS bf16 (fc_w row-major)
  unsigned char* STG = smem + 40960;       // 16  x FS bf16 ([z|embed] rows)
  unsigned char* HB  = smem + 44288;       // 2   x 16 x HS bf16 (h double buffer)

  const int tid  = threadIdx.x;
  const int wv   = tid >> 5;        // wave 0..3: owns hidden cols wv*16..wv*16+15
  const int lane = tid & 31;
  const int half = lane >> 4;
  const int lr   = lane & 15;
  const int rowBase = blockIdx.x * 16;   // 16 batch rows per workgroup

  // ---- stage weights / inputs into LDS as bf16 ----
  for (int e = tid; e < 192 * HID; e += 128) {
    int r = e >> 6, c = e & 63;
    *(unsigned short*)(WHH + (r * WS + c) * 2) = f2bf(w_hh[e]);
  }
  for (int e = tid; e < HID * 96; e += 128) {
    int r = e / 96, c = e - r * 96;
    *(unsigned short*)(FCW + (r * FS + c) * 2) = f2bf(fc_w[e]);
  }
  for (int e = tid; e < 16 * 96; e += 128) {
    int r = e / 96, c = e - r * 96;
    float v = (c < LAT) ? z[(rowBase + r) * LAT + c]
                        : embed_w[labels[rowBase + r] * HID + (c - LAT)];
    *(unsigned short*)(STG + (r * FS + c) * 2) = f2bf(v);
  }

  // ---- per-lane constants (this wave's hidden column = wv*16 + lr) ----
  const int   col = wv * 16 + lr;
  const float brh = 0.5f * (b_ih[col]       + b_hh[col]);       // r bias, pre-scaled
  const float bzh = 0.5f * (b_ih[HID + col] + b_hh[HID + col]); // z bias, pre-scaled
  const float bn1 = b_ih[2 * HID + col];                        // i_n
  const float bn2 = b_hh[2 * HID + col];                        // b_hh_n
  const float fcb = fc_b[col];
  const float ob  = (lr < 2) ? out_b[lr] : 0.0f;

  // ---- out_w^T B fragments (cols 0,1 = out features, rest zero) ----
  union { unsigned short h[16]; v16bf v; } bo0, bo1;
#pragma unroll
  for (int j = 0; j < 16; ++j) { bo0.h[j] = 0; bo1.h[j] = 0; }
  if (lr < 2) {
#pragma unroll
    for (int j = 0; j < 16; ++j) {
      bo0.h[j] = f2bf(out_w[lr * HID + (half * 16 + j)]);        // K 0..31
      bo1.h[j] = f2bf(out_w[lr * HID + (32 + half * 16 + j)]);   // K 32..63
    }
  }

  __syncthreads();

  // Load a v16bf fragment as two 16-byte chunks from an LDS row base pointer.
  auto ld_frag = [](const unsigned char* p, int off0, int off1) -> v16bf {
    union { uint4 u[2]; v16bf v; } f;
    f.u[0] = *(const uint4*)(p + off0);
    f.u[1] = *(const uint4*)(p + off1);
    return f.v;
  };

  // ---- resident w_hh B fragments for this wave's gate columns ----
  const unsigned char* bpR = WHH + (          wv * 16 + lr) * (WS * 2);
  const unsigned char* bpZ = WHH + (HID     + wv * 16 + lr) * (WS * 2);
  const unsigned char* bpN = WHH + (2 * HID + wv * 16 + lr) * (WS * 2);
  const int bOff0 = half * 32;        // B frag: 32 contiguous bytes per lane
  const int bOff1 = 64 + half * 32;   // second K block
  const v16bf bR0 = ld_frag(bpR, bOff0, bOff0 + 16);
  const v16bf bR1 = ld_frag(bpR, bOff1, bOff1 + 16);
  const v16bf bZ0 = ld_frag(bpZ, bOff0, bOff0 + 16);
  const v16bf bZ1 = ld_frag(bpZ, bOff1, bOff1 + 16);
  const v16bf bN0 = ld_frag(bpN, bOff0, bOff0 + 16);
  const v16bf bN1 = ld_frag(bpN, bOff1, bOff1 + 16);

  // ---- h0 = tanh([z|embed] @ fc_w^T + fc_b), K = 96 ----
  v8f acc = {};
  {
    const unsigned char* ap = STG + lr * (FS * 2);
    const unsigned char* bp = FCW + (wv * 16 + lr) * (FS * 2);
#pragma unroll
    for (int kb = 0; kb < 3; ++kb) {
      v16bf a = ld_frag(ap, kb * 64 + half * 32, kb * 64 + half * 32 + 16);
      v16bf b = ld_frag(bp, kb * 64 + half * 32, kb * 64 + half * 32 + 16);
      acc = WMMA_BF16(a, b, acc);
    }
  }
  float h[8];
#pragma unroll
  for (int k = 0; k < 8; ++k) h[k] = tanh_fast(acc[k] + fcb);

  // h publish base pointers (per-lane, both buffers)
  unsigned char* hpub0 = HB + ((half * 8) * HS + col) * 2;                  // buffer 0
  unsigned char* hpub1 = HB + (16 * HS * 2) + ((half * 8) * HS + col) * 2; // buffer 1

  // publish h0 into buffer 0
#pragma unroll
  for (int k = 0; k < 8; k += 2) {
    unsigned u = pack2bf(h[k], h[k + 1]);
    *(unsigned short*)(hpub0 + k * (HS * 2))       = (unsigned short)u;
    *(unsigned short*)(hpub0 + (k + 1) * (HS * 2)) = (unsigned short)(u >> 16);
  }
  __syncthreads();

  const unsigned char* aP0 = HB + lr * (HS * 2);                 // buffer 0
  const unsigned char* aP1 = HB + (16 * HS * 2) + lr * (HS * 2); // buffer 1
  const int aOff0 = half * 32;        // A frag: K0..31 chunks
  const int aOff1 = 64 + half * 32;   // A frag: K32..63 chunks

  v16bf a0 = ld_frag(aP0, aOff0, aOff0 + 16);
  v16bf a1 = ld_frag(aP0, aOff1, aOff1 + 16);

  // per-lane output base pointer (wave 0, lanes with lr<2 write)
  float* outp0 = out + (long)(rowBase + half * 8) * (SEQ * 2) + lr;

  // ---- one GRU step: consume a0/a1, publish into `hpub`, reload A from `aP` ----
  auto gru_step = [&](int s, unsigned char* hpub, const unsigned char* aP) {
    v8f aR = {}, aZ = {}, aN = {};
    aR = WMMA_BF16(a0, bR0, aR); aR = WMMA_BF16(a1, bR1, aR);
    aZ = WMMA_BF16(a0, bZ0, aZ); aZ = WMMA_BF16(a1, bZ1, aZ);
    aN = WMMA_BF16(a0, bN0, aN); aN = WMMA_BF16(a1, bN1, aN);

#pragma unroll
    for (int k = 0; k < 8; ++k) {
      // sigmoid(x) = 0.5*tanh(0.5x)+0.5 with bias folded into the pre-scale
      float r  = __builtin_fmaf(0.5f, tanh_fast(__builtin_fmaf(0.5f, aR[k], brh)), 0.5f);
      float zg = __builtin_fmaf(0.5f, tanh_fast(__builtin_fmaf(0.5f, aZ[k], bzh)), 0.5f);
      float n  = tanh_fast(__builtin_fmaf(r, aN[k] + bn2, bn1));
      h[k] = __builtin_fmaf(zg, h[k] - n, n);   // (1-z)*n + z*h
    }

#pragma unroll
    for (int k = 0; k < 8; k += 2) {
      unsigned u = pack2bf(h[k], h[k + 1]);
      *(unsigned short*)(hpub + k * (HS * 2))       = (unsigned short)u;
      *(unsigned short*)(hpub + (k + 1) * (HS * 2)) = (unsigned short)(u >> 16);
    }
    __syncthreads();

    a0 = ld_frag(aP, aOff0, aOff0 + 16);
    a1 = ld_frag(aP, aOff1, aOff1 + 16);

    // output projection: wave 0 only (wave-uniform branch, EXEC stays full)
    if (wv == 0) {
      v8f oa = {};
      oa = WMMA_BF16(a0, bo0.v, oa);
      oa = WMMA_BF16(a1, bo1.v, oa);
      if (lr < 2) {
#pragma unroll
        for (int k = 0; k < 8; ++k) {
          outp0[(long)k * (SEQ * 2) + s * 2] = oa[k] + ob;
        }
      }
    }
  };

  // ---- 180 sequential GRU steps, unrolled x2 so buffer pointers are static ----
  for (int s = 0; s < SEQ; s += 2) {
    gru_step(s,     hpub1, aP1);   // publish buf1, read A from buf1
    gru_step(s + 1, hpub0, aP0);   // publish buf0, read A from buf0
  }
}

extern "C" void kernel_launch(void* const* d_in, const int* in_sizes, int n_in,
                              void* d_out, int out_size, void* d_ws, size_t ws_size,
                              hipStream_t stream) {
  const float* z      = (const float*)d_in[0];
  const int*   labels = (const int*)d_in[1];
  const float* embed  = (const float*)d_in[2];
  const float* fc_w   = (const float*)d_in[3];
  const float* fc_b   = (const float*)d_in[4];
  // d_in[5] = w_ih (GRU input is all-zeros -> unused)
  const float* w_hh   = (const float*)d_in[6];
  const float* b_ih   = (const float*)d_in[7];
  const float* b_hh   = (const float*)d_in[8];
  const float* out_w  = (const float*)d_in[9];
  const float* out_b  = (const float*)d_in[10];

  gru_gen_kernel<<<B_BATCH / 16, 128, 0, stream>>>(
      z, labels, embed, fc_w, fc_b, w_hh, b_ih, b_hh, out_w, out_b,
      (float*)d_out);
}
